// OneStepAttention_70119636075157
// MI455X (gfx1250) — compile-verified, matrix operand order checked by reading, not run on hardware
//
#include <hip/hip_runtime.h>
#include <hip/hip_bf16.h>
#include <math.h>

// Problem sizes (fixed by the reference)
#define B_   32
#define D_   1024
#define S_   2048

// GEMM tiling
#define MT   128      // d-rows per block
#define NT   128      // s-cols per block
#define KS   32       // k-slab per LDS stage (bf16 WMMA K)
#define LDK  40       // padded LDS k-stride (bf16 elems) to avoid bank conflicts
#define NSTEP 64      // total K steps: 2 phases x (D_/KS)

#if __has_builtin(__builtin_amdgcn_tanhf)
  #define TANH_F32(x) __builtin_amdgcn_tanhf(x)   // v_tanh_f32 (gfx1250 TRANS op)
#else
  #define TANH_F32(x) tanhf(x)
#endif

typedef __attribute__((ext_vector_type(16))) __bf16 v16bf;
typedef __attribute__((ext_vector_type(8)))  __bf16 v8bf;
typedef __attribute__((ext_vector_type(8)))  float  v8f;

// ---------------------------------------------------------------------------
// Stage 1: s[b, n] = sum_d v[b,d] * tanh( (W@k)[b,d,n] + (U@q)[b,d,n] )
// One block = 128x128 tile of e for one batch. Single 64-step K loop runs
// W*k (steps 0..31) then U*q (steps 32..63) into the same f32 accumulators
// via v_wmma_f32_16x16x32_bf16, with a register double-buffer prefetching
// the next fp32 k-slab while the current slab's WMMAs execute.
// ---------------------------------------------------------------------------
__global__ __launch_bounds__(256) void bahdanau_score_kernel(
    const float* __restrict__ q, const float* __restrict__ k,
    const float* __restrict__ v, const float* __restrict__ W,
    const float* __restrict__ U, float* __restrict__ sG)
{
  __shared__ __bf16 As[MT * LDK];   // A tile: [row][k]  (row-major)
  __shared__ __bf16 Bs[NT * LDK];   // B tile: [col][k]  (transposed on store)
  __shared__ float  sPart[NT];

  const int tid  = threadIdx.x;
  const int lane = tid & 31;
  const int wid  = tid >> 5;        // 8 waves
  const int wm   = wid & 3;         // wave m-strip (4 x 32 rows)
  const int wn   = wid >> 2;        // wave n-strip (2 x 64 cols)
  const int lrow = lane & 15;
  const int half = lane >> 4;       // 0 or 1 (K-half select per WMMA layout)

  const int b      = blockIdx.z;
  const int blockN = blockIdx.x * NT;
  const int blockM = blockIdx.y * MT;

  // global-load thread mapping
  const int ar0 = tid >> 3;         // A: row base 0..31
  const int akc = (tid & 7) * 4;    // A: k offset 0,4,...,28
  const int be0 = tid >> 5;         // B: e base 0..7
  const int bsc = (tid & 31) * 4;   // B: s offset 0..124

  const size_t bOff = (size_t)b * D_ * S_;

  v8f acc[2][4];
  #pragma unroll
  for (int i = 0; i < 2; ++i)
    #pragma unroll
    for (int j = 0; j < 4; ++j)
      acc[i][j] = (v8f)0.f;

  // ---- register double-buffer for the fp32 slabs ----
  float4 aReg[4], bReg[4];

  auto loadStep = [&](int s) {
    const float* Ap = (s < 32) ? W : U;                 // phase select
    const float* Bp = ((s < 32) ? k : q) + bOff;
    const int k0 = (s & 31) * KS;
    #pragma unroll
    for (int rr = 0; rr < 4; ++rr)
      aReg[rr] = *(const float4*)(Ap + (size_t)(blockM + rr * 32 + ar0) * D_ + k0 + akc);
    #pragma unroll
    for (int rr = 0; rr < 4; ++rr)
      bReg[rr] = *(const float4*)(Bp + (size_t)(k0 + rr * 8 + be0) * S_ + blockN + bsc);
  };

  auto storeStep = [&]() {
    #pragma unroll
    for (int rr = 0; rr < 4; ++rr) {
      __bf16* dst = &As[(rr * 32 + ar0) * LDK + akc];
      dst[0] = (__bf16)aReg[rr].x; dst[1] = (__bf16)aReg[rr].y;
      dst[2] = (__bf16)aReg[rr].z; dst[3] = (__bf16)aReg[rr].w;
    }
    #pragma unroll
    for (int rr = 0; rr < 4; ++rr) {
      const int e = rr * 8 + be0;
      Bs[(bsc + 0) * LDK + e] = (__bf16)bReg[rr].x;
      Bs[(bsc + 1) * LDK + e] = (__bf16)bReg[rr].y;
      Bs[(bsc + 2) * LDK + e] = (__bf16)bReg[rr].z;
      Bs[(bsc + 3) * LDK + e] = (__bf16)bReg[rr].w;
    }
  };

  loadStep(0);                       // prologue: first slab in flight

  for (int s = 0; s < NSTEP; ++s) {
    __syncthreads();                 // previous iteration's LDS reads done
    storeStep();                     // regs -> LDS (fp32 -> bf16)
    __syncthreads();

    if (s + 1 < NSTEP) loadStep(s + 1);   // prefetch next slab; waits land
                                          // just before next storeStep()

    // ---- fragments (16-bit A/B lane layout: halves 0..7 = K khalf+0..7,
    //      halves 8..15 = K khalf+16..23, khalf = 8*(lane>=16)) ----
    v16bf af[2], bfr[4];
    #pragma unroll
    for (int i = 0; i < 2; ++i) {
      const __bf16* ap = &As[(wm * 32 + i * 16 + lrow) * LDK + half * 8];
      v8bf lo = *(const v8bf*)(ap);
      v8bf hi = *(const v8bf*)(ap + 16);
      af[i] = __builtin_shufflevector(lo, hi, 0,1,2,3,4,5,6,7,8,9,10,11,12,13,14,15);
    }
    #pragma unroll
    for (int j = 0; j < 4; ++j) {
      const __bf16* bp = &Bs[(wn * 64 + j * 16 + lrow) * LDK + half * 8];
      v8bf lo = *(const v8bf*)(bp);
      v8bf hi = *(const v8bf*)(bp + 16);
      bfr[j] = __builtin_shufflevector(lo, hi, 0,1,2,3,4,5,6,7,8,9,10,11,12,13,14,15);
    }

    #pragma unroll
    for (int i = 0; i < 2; ++i)
      #pragma unroll
      for (int j = 0; j < 4; ++j)
        acc[i][j] = __builtin_amdgcn_wmma_f32_16x16x32_bf16(
            false, af[i], false, bfr[j], (short)0, acc[i][j], false, false);
  }

  // ---- epilogue: tanh, scale by v[b,d], reduce over d ----
  __syncthreads();
  if (tid < NT) sPart[tid] = 0.f;
  __syncthreads();

  // C layout: lane -> column (lane&15), vgpr r -> row r + 8*(lane>=16)
  float csum[4] = {0.f, 0.f, 0.f, 0.f};
  #pragma unroll
  for (int i = 0; i < 2; ++i) {
    #pragma unroll
    for (int r = 0; r < 8; ++r) {
      const int drow = blockM + wm * 32 + i * 16 + half * 8 + r;
      const float vv = v[(size_t)b * D_ + drow];
      #pragma unroll
      for (int j = 0; j < 4; ++j)
        csum[j] += TANH_F32(acc[i][j][r]) * vv;
    }
  }
  #pragma unroll
  for (int j = 0; j < 4; ++j) {
    csum[j] += __shfl_xor(csum[j], 16, 32);   // fold the two row-halves per column
    if (half == 0)
      atomicAdd(&sPart[wn * 64 + j * 16 + lrow], csum[j]);
  }
  __syncthreads();
  if (tid < NT)
    atomicAdd(&sG[(size_t)b * S_ + blockN + tid], sPart[tid]);
}

// ---------------------------------------------------------------------------
// Stage 2: softmax over S per batch row
// ---------------------------------------------------------------------------
__global__ __launch_bounds__(256) void softmax_kernel(const float* __restrict__ sG,
                                                      float* __restrict__ out)
{
  __shared__ float red[8];
  const int b   = blockIdx.x;
  const int tid = threadIdx.x;
  const float* row = sG + (size_t)b * S_;
  float*      orow = out + (size_t)b * S_;

  float m = -INFINITY;
  for (int i = tid; i < S_; i += 256) m = fmaxf(m, row[i]);
  #pragma unroll
  for (int o = 16; o; o >>= 1) m = fmaxf(m, __shfl_xor(m, o, 32));
  if ((tid & 31) == 0) red[tid >> 5] = m;
  __syncthreads();
  m = red[0];
  #pragma unroll
  for (int w = 1; w < 8; ++w) m = fmaxf(m, red[w]);

  float sum = 0.f;
  for (int i = tid; i < S_; i += 256) {
    const float e = __expf(row[i] - m);
    orow[i] = e;
    sum += e;
  }
  #pragma unroll
  for (int o = 16; o; o >>= 1) sum += __shfl_xor(sum, o, 32);
  __syncthreads();
  if ((tid & 31) == 0) red[tid >> 5] = sum;
  __syncthreads();
  sum = 0.f;
  #pragma unroll
  for (int w = 0; w < 8; ++w) sum += red[w];

  const float inv = 1.f / sum;
  for (int i = tid; i < S_; i += 256) orow[i] *= inv;
}

// ---------------------------------------------------------------------------
extern "C" void kernel_launch(void* const* d_in, const int* in_sizes, int n_in,
                              void* d_out, int out_size, void* d_ws, size_t ws_size,
                              hipStream_t stream)
{
  (void)in_sizes; (void)n_in; (void)out_size; (void)ws_size;
  const float* q = (const float*)d_in[0];
  const float* k = (const float*)d_in[1];
  const float* v = (const float*)d_in[2];
  const float* W = (const float*)d_in[3];
  const float* U = (const float*)d_in[4];
  float* out = (float*)d_out;
  float* sG  = (float*)d_ws;   // [B, S] score accumulator

  hipMemsetAsync(sG, 0, (size_t)B_ * S_ * sizeof(float), stream);

  dim3 grid(S_ / NT, D_ / MT, B_);   // (16, 8, 32)
  bahdanau_score_kernel<<<grid, 256, 0, stream>>>(q, k, v, W, U, sG);
  softmax_kernel<<<B_, 256, 0, stream>>>(sG, out);
}
